// decoder_8589934592373
// MI455X (gfx1250) — compile-verified, matrix operand order checked by reading, not run on hardware
//
#include <hip/hip_runtime.h>
#include <hip/hip_bf16.h>
#include <math.h>

#define S_LEN   1024
#define DMODEL  1024
#define NHEAD   16
#define HDIM    64
#define NLAYER  4
#define NCLASS  1000
#define FFDIM   4096
#define PREL    2047          // S + S - 1 relative positions
#define PSTR    2048          // padded stride for Rrel/Bq
#define EPS_LN  1e-5f

typedef __attribute__((ext_vector_type(16))) __bf16       v16bf;
typedef __attribute__((ext_vector_type(8)))  float        v8f;
typedef __attribute__((ext_vector_type(4)))  unsigned int u32x4;

union FragU { u32x4 q[2]; v16bf v; };

__device__ __forceinline__ unsigned short f2bf(float f) {
    unsigned int u = __float_as_uint(f);
    u += 0x7FFFu + ((u >> 16) & 1u);      // round-to-nearest-even
    return (unsigned short)(u >> 16);
}

// ---------------------------------------------------------------------------
// fp32 -> bf16 staging kernels (run once per GEMM operand, outside hot loop)
// ---------------------------------------------------------------------------
__global__ __launch_bounds__(256)
void cvt_copy_bf16(const float* __restrict__ src, int ld,
                   unsigned short* __restrict__ dst, int ldd,
                   int rows, int cols)
{
    int idx = blockIdx.x * 256 + threadIdx.x;
    if (idx >= rows * cols) return;
    int r = idx / cols, c = idx % cols;
    dst[(size_t)r * ldd + c] = f2bf(src[(size_t)r * ld + c]);
}

// dst[c][r] = src[r][c]  : fp32 [R][C] (row stride ld) -> bf16 [C][R] (row stride ldd)
__global__ __launch_bounds__(256)
void cvt_transpose_bf16(const float* __restrict__ src, int ld,
                        unsigned short* __restrict__ dst, int ldd,
                        int R, int C)
{
    int idx = blockIdx.x * 256 + threadIdx.x;
    if (idx >= R * C) return;
    int c = idx / R, r = idx % R;                 // writes coalesced over r
    dst[(size_t)c * ldd + r] = f2bf(src[(size_t)r * ld + c]);
}

// ---------------------------------------------------------------------------
// GEMM: C[M,N] = Abf[M,K] @ Bbf[N,K]^T (+bias, +ELU).  A,B are bf16, both
// stored row-major along K.  Tiles 128x128x32, double-buffered LDS filled by
// global_load_async_to_lds_b128 (ASYNCcnt), fragments via ds_load_b128,
// compute via v_wmma_f32_16x16x32_bf16.
// Requirements (guaranteed by callers): M %128==0 (=1024), K %32==0,
// lda/ldb %8==0, B allocation has >= ceil(N/128)*128 valid-to-read rows.
// ---------------------------------------------------------------------------
#define TILE_M 128
#define TILE_N 128
#define TILE_K 32
#define LDS_H  40             // halfs per LDS row (32 + 8 pad) = 80B, 16B aligned
#define GF_ELU 2

__device__ __forceinline__ v16bf load_frag(const unsigned short* row, int g) {
    FragU f;
    const char* p = (const char*)row;
    f.q[0] = *(const u32x4*)(p + 16 * g);        // halfs K = 8g .. 8g+7
    f.q[1] = *(const u32x4*)(p + 32 + 16 * g);   // halfs K = 16+8g .. 16+8g+7
    return f.v;
}

__global__ __launch_bounds__(256)
void gemm_bf16_wmma(const unsigned short* __restrict__ A, int lda,
                    const unsigned short* __restrict__ B, int ldb,
                    float* __restrict__ C, int ldc,
                    int M, int N, int K,
                    const float* __restrict__ bias, int flags)
{
    __shared__ unsigned short As[2][TILE_M * LDS_H];
    __shared__ unsigned short Bs[2][TILE_N * LDS_H];

    const int tid  = threadIdx.x;
    const int lane = tid & 31;
    const int wav  = tid >> 5;
    const int wrow = wav & 3;         // 4 wave-rows * 32 rows
    const int wcol = wav >> 2;        // 2 wave-cols * 64 cols
    const int lo   = lane & 15;
    const int g    = lane >> 4;

    const int m0 = blockIdx.y * TILE_M;
    const int n0 = blockIdx.x * TILE_N;

    // each thread DMAs two 16B chunks per matrix per stage:
    // rows r0 and r0+64, same 8-half column group
    const int r0 = tid >> 2;
    const int c8 = (tid & 3) << 3;
    const unsigned ldsA[2] = { (unsigned)(unsigned long long)&As[0][0],
                               (unsigned)(unsigned long long)&As[1][0] };
    const unsigned ldsB[2] = { (unsigned)(unsigned long long)&Bs[0][0],
                               (unsigned)(unsigned long long)&Bs[1][0] };
    const unsigned ldsOff0 = (unsigned)((r0 * LDS_H + c8) * 2);
    const unsigned ldsOff1 = (unsigned)(((r0 + 64) * LDS_H + c8) * 2);

    auto issue_stage = [&](int k0, int sel) {
        unsigned long long ga0 = (unsigned long long)(A + (size_t)(m0 + r0)      * lda + k0 + c8);
        unsigned long long ga1 = (unsigned long long)(A + (size_t)(m0 + r0 + 64) * lda + k0 + c8);
        unsigned long long gb0 = (unsigned long long)(B + (size_t)(n0 + r0)      * ldb + k0 + c8);
        unsigned long long gb1 = (unsigned long long)(B + (size_t)(n0 + r0 + 64) * ldb + k0 + c8);
        asm volatile("global_load_async_to_lds_b128 %0, %1, off"
                     :: "v"(ldsA[sel] + ldsOff0), "v"(ga0) : "memory");
        asm volatile("global_load_async_to_lds_b128 %0, %1, off"
                     :: "v"(ldsA[sel] + ldsOff1), "v"(ga1) : "memory");
        asm volatile("global_load_async_to_lds_b128 %0, %1, off"
                     :: "v"(ldsB[sel] + ldsOff0), "v"(gb0) : "memory");
        asm volatile("global_load_async_to_lds_b128 %0, %1, off"
                     :: "v"(ldsB[sel] + ldsOff1), "v"(gb1) : "memory");
    };

    v8f acc[2][4];
#pragma unroll
    for (int i = 0; i < 2; ++i)
#pragma unroll
        for (int j = 0; j < 4; ++j)
#pragma unroll
            for (int e = 0; e < 8; ++e) acc[i][j][e] = 0.f;

    const int nsteps = K / TILE_K;
    issue_stage(0, 0);                               // prologue

    for (int t = 0; t < nsteps; ++t) {
        const int cur = t & 1;
        asm volatile("s_wait_asynccnt 0x0" ::: "memory");
        __syncthreads();
        if (t + 1 < nsteps) issue_stage((t + 1) * TILE_K, cur ^ 1);

        const unsigned short* Asc = &As[cur][0];
        const unsigned short* Bsc = &Bs[cur][0];
        v16bf afr[2], bfr[4];
#pragma unroll
        for (int i = 0; i < 2; ++i)
            afr[i] = load_frag(Asc + (wrow * 32 + i * 16 + lo) * LDS_H, g);
#pragma unroll
        for (int j = 0; j < 4; ++j)
            bfr[j] = load_frag(Bsc + (wcol * 64 + j * 16 + lo) * LDS_H, g);
#pragma unroll
        for (int i = 0; i < 2; ++i)
#pragma unroll
            for (int j = 0; j < 4; ++j)
                acc[i][j] = __builtin_amdgcn_wmma_f32_16x16x32_bf16(
                    false, afr[i], false, bfr[j], (short)0, acc[i][j], false, false);
    }

    // ---- epilogue ----
#pragma unroll
    for (int i = 0; i < 2; ++i)
#pragma unroll
        for (int j = 0; j < 4; ++j) {
            int col = n0 + wcol * 64 + j * 16 + lo;
            if (col >= N) continue;
            float bv = bias ? bias[col] : 0.f;
#pragma unroll
            for (int r = 0; r < 8; ++r) {
                int row = m0 + wrow * 32 + i * 16 + g * 8 + r;
                if (row >= M) continue;
                float x = acc[i][j][r] + bv;
                if (flags & GF_ELU) x = (x > 0.f) ? x : (__expf(x) - 1.f);
                C[(size_t)row * ldc + col] = x;
            }
        }
}

// ---------------------------------------------------------------------------
// Relative-position table Rrel[d][p], p = (q-k)+1023, all PSTR columns filled.
// ---------------------------------------------------------------------------
__global__ void rel_fill(float* __restrict__ R) {
    int idx = blockIdx.x * blockDim.x + threadIdx.x;
    if (idx >= HDIM * PSTR) return;
    int d = idx / PSTR, p = idx % PSTR;
    float rel = (float)(p - (S_LEN - 1));
    float fr  = __powf(10000.f, -(float)(d >> 1) / 32.f);
    float ang = rel * fr;
    R[(size_t)d * PSTR + p] = (d & 1) ? __cosf(ang) : __sinf(ang);
}

// vb[h][p] = sum_d v[h,d] * Rrel[d,p]
__global__ void vb_fill(const float* __restrict__ v, const float* __restrict__ R,
                        float* __restrict__ vb) {
    int idx = blockIdx.x * blockDim.x + threadIdx.x;
    if (idx >= NHEAD * PREL) return;
    int h = idx / PREL, p = idx % PREL;
    float s = 0.f;
    for (int d = 0; d < HDIM; ++d) s += v[h * HDIM + d] * R[(size_t)d * PSTR + p];
    vb[(size_t)h * PSTR + p] = s;
}

// ub[h][k] = sum_d u[h,d] * K[k, h*64+d]
__global__ void ub_fill(const float* __restrict__ u, const float* __restrict__ kmat,
                        float* __restrict__ ub) {
    int idx = blockIdx.x * blockDim.x + threadIdx.x;
    if (idx >= NHEAD * S_LEN) return;
    int h = idx / S_LEN, kk = idx % S_LEN;
    float s = 0.f;
    for (int d = 0; d < HDIM; ++d) s += u[h * HDIM + d] * kmat[(size_t)kk * DMODEL + h * HDIM + d];
    ub[(size_t)h * S_LEN + kk] = s;
}

// ---------------------------------------------------------------------------
// Score finalize (+rel-pos gather, u/v bias, causal mask) + row softmax.
// ---------------------------------------------------------------------------
__global__ __launch_bounds__(256)
void attn_softmax(float* __restrict__ scores, const float* __restrict__ Bq,
                  const float* __restrict__ ubh, const float* __restrict__ vbh,
                  int nk, int causal)
{
    __shared__ float red[256];
    int q = blockIdx.x;
    float* row = scores + (size_t)q * nk;
    const float* bq = Bq + (size_t)q * PSTR;
    int off = nk - 1;
    float lmax = -3.4e38f;
    for (int k = threadIdx.x; k < nk; k += 256) {
        int p = q - k + off;
        float s = row[k] + bq[p] + ubh[k] + vbh[p];
        if (causal && k > q) s = -1e18f;
        row[k] = s;
        lmax = fmaxf(lmax, s);
    }
    red[threadIdx.x] = lmax; __syncthreads();
    for (int st = 128; st > 0; st >>= 1) {
        if (threadIdx.x < st) red[threadIdx.x] = fmaxf(red[threadIdx.x], red[threadIdx.x + st]);
        __syncthreads();
    }
    float m = red[0]; __syncthreads();
    float lsum = 0.f;
    for (int k = threadIdx.x; k < nk; k += 256) {
        float e = __expf(row[k] - m);
        row[k] = e; lsum += e;
    }
    red[threadIdx.x] = lsum; __syncthreads();
    for (int st = 128; st > 0; st >>= 1) {
        if (threadIdx.x < st) red[threadIdx.x] += red[threadIdx.x + st];
        __syncthreads();
    }
    float inv = 1.f / red[0];
    for (int k = threadIdx.x; k < nk; k += 256) row[k] *= inv;
}

// ---------------------------------------------------------------------------
// out[r] = LN(a[r] + b[r]) * gamma + beta   (D = 1024, one block per row)
// ---------------------------------------------------------------------------
__global__ __launch_bounds__(256)
void ln_residual(const float* __restrict__ a, const float* __restrict__ b,
                 const float* __restrict__ gamma, const float* __restrict__ beta,
                 float* __restrict__ out)
{
    __shared__ float red[256];
    int r = blockIdx.x;
    float x[4];
    float s = 0.f;
#pragma unroll
    for (int i = 0; i < 4; ++i) {
        int c = threadIdx.x + i * 256;
        x[i] = a[(size_t)r * DMODEL + c] + b[(size_t)r * DMODEL + c];
        s += x[i];
    }
    red[threadIdx.x] = s; __syncthreads();
    for (int st = 128; st > 0; st >>= 1) {
        if (threadIdx.x < st) red[threadIdx.x] += red[threadIdx.x + st];
        __syncthreads();
    }
    float mean = red[0] * (1.f / DMODEL); __syncthreads();
    float s2 = 0.f;
#pragma unroll
    for (int i = 0; i < 4; ++i) { float d = x[i] - mean; s2 += d * d; }
    red[threadIdx.x] = s2; __syncthreads();
    for (int st = 128; st > 0; st >>= 1) {
        if (threadIdx.x < st) red[threadIdx.x] += red[threadIdx.x + st];
        __syncthreads();
    }
    float rstd = rsqrtf(red[0] * (1.f / DMODEL) + EPS_LN);
#pragma unroll
    for (int i = 0; i < 4; ++i) {
        int c = threadIdx.x + i * 256;
        out[(size_t)r * DMODEL + c] = (x[i] - mean) * rstd * gamma[c] + beta[c];
    }
}

// ---------------------------------------------------------------------------
// Final head: softmax(buf[last] @ Wout + bout) over 1000 classes. One block.
// ---------------------------------------------------------------------------
__global__ __launch_bounds__(1024)
void head_softmax(const float* __restrict__ buf, const float* __restrict__ Wout,
                  const float* __restrict__ bout, float* __restrict__ out)
{
    __shared__ float red[1024];
    int c = threadIdx.x;
    float val = -3.4e38f;
    if (c < NCLASS) {
        float s = bout[c];
        const float* x = buf + (size_t)(S_LEN - 1) * DMODEL;
        for (int d = 0; d < DMODEL; ++d) s += x[d] * Wout[(size_t)d * NCLASS + c];
        val = s;
    }
    red[c] = val; __syncthreads();
    for (int st = 512; st > 0; st >>= 1) {
        if (c < st) red[c] = fmaxf(red[c], red[c + st]);
        __syncthreads();
    }
    float m = red[0]; __syncthreads();
    float e = (c < NCLASS) ? __expf(val - m) : 0.f;
    red[c] = e; __syncthreads();
    for (int st = 512; st > 0; st >>= 1) {
        if (c < st) red[c] += red[c + st];
        __syncthreads();
    }
    if (c < NCLASS) out[c] = e / red[0];
}

// ---------------------------------------------------------------------------
// Host-side orchestration
// ---------------------------------------------------------------------------
struct Bufs {
    float *qb, *kb, *vb, *attn, *scores, *Bq, *Rrel, *ub, *vbt;
    unsigned short *Xbf, *Qbf, *Kbf, *Vt, *Pbf, *Wt, *Rbf;
};

static inline void launch_gemm(hipStream_t s, const unsigned short* A, int lda,
                               const unsigned short* B, int ldb, float* C, int ldc,
                               int M, int N, int K, const float* bias, int flags) {
    dim3 grid((N + TILE_N - 1) / TILE_N, (M + TILE_M - 1) / TILE_M);
    gemm_bf16_wmma<<<grid, dim3(256), 0, s>>>(A, lda, B, ldb, C, ldc, M, N, K, bias, flags);
}
static inline void cvt_copy(hipStream_t s, const float* src, int ld,
                            unsigned short* dst, int ldd, int rows, int cols) {
    int n = rows * cols;
    cvt_copy_bf16<<<(n + 255) / 256, 256, 0, s>>>(src, ld, dst, ldd, rows, cols);
}
static inline void cvt_tr(hipStream_t s, const float* src, int ld,
                          unsigned short* dst, int ldd, int R, int C) {
    int n = R * C;
    cvt_transpose_bf16<<<(n + 255) / 256, 256, 0, s>>>(src, ld, dst, ldd, R, C);
}

// aQ/aK/aV: pre-converted bf16 [S][D] activations feeding the projections.
static void run_attention(hipStream_t s, const Bufs& w,
                          const unsigned short* aQ, const unsigned short* aK,
                          const unsigned short* aV,
                          const float* Wq, const float* Wk, const float* Wv,
                          const float* Wo, const float* bo,
                          const float* u, const float* vbias, int causal,
                          float* out)
{
    cvt_tr(s, Wq, DMODEL, w.Wt, DMODEL, DMODEL, DMODEL);
    launch_gemm(s, aQ, DMODEL, w.Wt, DMODEL, w.qb, DMODEL, S_LEN, DMODEL, DMODEL, nullptr, 0);
    cvt_tr(s, Wk, DMODEL, w.Wt, DMODEL, DMODEL, DMODEL);
    launch_gemm(s, aK, DMODEL, w.Wt, DMODEL, w.kb, DMODEL, S_LEN, DMODEL, DMODEL, nullptr, 0);
    cvt_tr(s, Wv, DMODEL, w.Wt, DMODEL, DMODEL, DMODEL);
    launch_gemm(s, aV, DMODEL, w.Wt, DMODEL, w.vb, DMODEL, S_LEN, DMODEL, DMODEL, nullptr, 0);

    ub_fill<<<(NHEAD * S_LEN + 255) / 256, 256, 0, s>>>(u, w.kb, w.ub);
    vb_fill<<<(NHEAD * PREL + 255) / 256, 256, 0, s>>>(vbias, w.Rrel, w.vbt);

    cvt_copy(s, w.qb, DMODEL, w.Qbf, DMODEL, S_LEN, DMODEL);
    cvt_copy(s, w.kb, DMODEL, w.Kbf, DMODEL, S_LEN, DMODEL);
    cvt_tr(s, w.vb, DMODEL, w.Vt, S_LEN, S_LEN, DMODEL);   // Vt[d][k] = v[k][d]

    for (int h = 0; h < NHEAD; ++h) {
        // content scores: q_h @ k_h^T  (both [n][K] along features)
        launch_gemm(s, w.Qbf + h * HDIM, DMODEL, w.Kbf + h * HDIM, DMODEL,
                    w.scores, S_LEN, S_LEN, S_LEN, HDIM, nullptr, 0);
        // content-position: q_h @ Rrel  -> Bq[q][p]
        launch_gemm(s, w.Qbf + h * HDIM, DMODEL, w.Rbf, HDIM,
                    w.Bq, PSTR, S_LEN, PREL, HDIM, nullptr, 0);
        attn_softmax<<<S_LEN, 256, 0, s>>>(w.scores, w.Bq, w.ub + h * S_LEN,
                                           w.vbt + (size_t)h * PSTR, S_LEN, causal);
        // probs @ v_h : B rows = output features (Vt slice), K = key index
        cvt_copy(s, w.scores, S_LEN, w.Pbf, S_LEN, S_LEN, S_LEN);
        launch_gemm(s, w.Pbf, S_LEN, w.Vt + (size_t)h * HDIM * S_LEN, S_LEN,
                    w.attn + h * HDIM, DMODEL, S_LEN, HDIM, S_LEN, nullptr, 0);
    }
    cvt_copy(s, w.attn, DMODEL, w.Xbf, DMODEL, S_LEN, DMODEL);
    cvt_tr(s, Wo, DMODEL, w.Wt, DMODEL, DMODEL, DMODEL);
    launch_gemm(s, w.Xbf, DMODEL, w.Wt, DMODEL, out, DMODEL, S_LEN, DMODEL, DMODEL, bo, 0);
}

extern "C" void kernel_launch(void* const* d_in, const int* in_sizes, int n_in,
                              void* d_out, int out_size, void* d_ws, size_t ws_size,
                              hipStream_t stream)
{
    (void)in_sizes; (void)n_in; (void)out_size; (void)ws_size;
    const float* seq   = (const float*)d_in[0];
    const float* encv  = (const float*)d_in[1];
    const float* enck  = (const float*)d_in[2];
    const float* Wq_s  = (const float*)d_in[3];
    const float* Wk_s  = (const float*)d_in[4];
    const float* Wv_s  = (const float*)d_in[5];
    const float* Wo_s  = (const float*)d_in[6];
    const float* bo_s  = (const float*)d_in[7];
    const float* u_s   = (const float*)d_in[8];
    const float* v_s   = (const float*)d_in[9];
    const float* gn    = (const float*)d_in[10];
    const float* bn    = (const float*)d_in[11];
    const float* Wq_c  = (const float*)d_in[12];
    const float* Wk_c  = (const float*)d_in[13];
    const float* Wv_c  = (const float*)d_in[14];
    const float* Wo_c  = (const float*)d_in[15];
    const float* bo_c  = (const float*)d_in[16];
    const float* u_c   = (const float*)d_in[17];
    const float* v_c   = (const float*)d_in[18];
    const float* g0    = (const float*)d_in[19];
    const float* b0    = (const float*)d_in[20];
    const float* g1    = (const float*)d_in[21];
    const float* b1    = (const float*)d_in[22];
    const float* W1    = (const float*)d_in[23];
    const float* bf1   = (const float*)d_in[24];
    const float* W2    = (const float*)d_in[25];
    const float* bf2   = (const float*)d_in[26];
    const float* Wout  = (const float*)d_in[27];
    const float* bout  = (const float*)d_in[28];

    float* ws = (float*)d_ws;
    const size_t MB = 1u << 20;              // 1M floats
    float* buf    = ws + 0 * MB;             // S*D
    float* t0     = ws + 1 * MB;
    float* t1     = ws + 2 * MB;
    Bufs w;
    w.qb     = ws + 3 * MB;
    w.kb     = ws + 4 * MB;
    w.vb     = ws + 5 * MB;
    w.attn   = ws + 6 * MB;
    float* proj = ws + 7 * MB;
    float* ff   = ws + 8 * MB;               // S*FFDIM (4M floats)
    w.scores = ws + 12 * MB;                 // S*S
    w.Bq     = ws + 13 * MB;                 // S*PSTR (2M)
    w.Rrel   = ws + 15 * MB;                 // HDIM*PSTR = 131072
    w.ub     = w.Rrel + (size_t)HDIM * PSTR; // 16384
    w.vbt    = w.ub + (size_t)NHEAD * S_LEN; // 32768

    // bf16 staging region starts at 64 MB
    unsigned short* hb = (unsigned short*)(ws + 16 * MB);
    size_t ho = 0;
    auto halloc = [&](size_t n) { unsigned short* p = hb + ho; ho += (n + 63) & ~(size_t)63; return p; };
    w.Xbf  = halloc((size_t)S_LEN * DMODEL);
    w.Qbf  = halloc((size_t)S_LEN * DMODEL);
    w.Kbf  = halloc((size_t)S_LEN * DMODEL);
    w.Vt   = halloc((size_t)(DMODEL + 128) * S_LEN);   // slack rows for 128-row tiles
    w.Pbf  = halloc((size_t)S_LEN * S_LEN);
    w.Wt   = halloc((size_t)FFDIM * DMODEL);           // largest transposed weight
    unsigned short* Ekbf = halloc((size_t)S_LEN * DMODEL);
    unsigned short* Evbf = halloc((size_t)S_LEN * DMODEL);
    w.Rbf  = halloc((size_t)PSTR * HDIM);
    unsigned short* FFbf = halloc((size_t)S_LEN * FFDIM);

    hipMemcpyAsync(buf, seq, (size_t)S_LEN * DMODEL * sizeof(float),
                   hipMemcpyDeviceToDevice, stream);
    rel_fill<<<(HDIM * PSTR + 255) / 256, 256, 0, stream>>>(w.Rrel);
    cvt_tr(stream, w.Rrel, PSTR, w.Rbf, HDIM, HDIM, PSTR);   // Rbf[p][d]
    cvt_copy(stream, enck, DMODEL, Ekbf, DMODEL, S_LEN, DMODEL);
    cvt_copy(stream, encv, DMODEL, Evbf, DMODEL, S_LEN, DMODEL);

    for (int l = 0; l < NLAYER; ++l) {
        // --- masked relative self-attention + residual LN ---
        cvt_copy(stream, buf, DMODEL, w.Xbf, DMODEL, S_LEN, DMODEL);
        run_attention(stream, w, w.Xbf, w.Xbf, w.Xbf,
                      Wq_s + (size_t)l * DMODEL * DMODEL, Wk_s + (size_t)l * DMODEL * DMODEL,
                      Wv_s + (size_t)l * DMODEL * DMODEL, Wo_s + (size_t)l * DMODEL * DMODEL,
                      bo_s + (size_t)l * DMODEL, u_s + (size_t)l * NHEAD * HDIM,
                      v_s + (size_t)l * NHEAD * HDIM, /*causal=*/1, proj);
        ln_residual<<<S_LEN, 256, 0, stream>>>(proj, buf, gn + (size_t)l * DMODEL,
                                               bn + (size_t)l * DMODEL, t0);
        // --- cross-attention over encoder memory + residual LN ---
        cvt_copy(stream, t0, DMODEL, w.Xbf, DMODEL, S_LEN, DMODEL);
        run_attention(stream, w, w.Xbf, Ekbf, Evbf,
                      Wq_c + (size_t)l * DMODEL * DMODEL, Wk_c + (size_t)l * DMODEL * DMODEL,
                      Wv_c + (size_t)l * DMODEL * DMODEL, Wo_c + (size_t)l * DMODEL * DMODEL,
                      bo_c + (size_t)l * DMODEL, u_c + (size_t)l * NHEAD * HDIM,
                      v_c + (size_t)l * NHEAD * HDIM, /*causal=*/0, proj);
        ln_residual<<<S_LEN, 256, 0, stream>>>(proj, t0, g0 + (size_t)l * DMODEL,
                                               b0 + (size_t)l * DMODEL, t1);
        // --- feed-forward (ELU) + residual LN ---
        cvt_copy(stream, t1, DMODEL, w.Xbf, DMODEL, S_LEN, DMODEL);
        cvt_tr(stream, W1 + (size_t)l * DMODEL * FFDIM, FFDIM, w.Wt, DMODEL, DMODEL, FFDIM);
        launch_gemm(stream, w.Xbf, DMODEL, w.Wt, DMODEL, ff, FFDIM,
                    S_LEN, FFDIM, DMODEL, bf1 + (size_t)l * FFDIM, GF_ELU);
        cvt_copy(stream, ff, FFDIM, FFbf, FFDIM, S_LEN, FFDIM);
        cvt_tr(stream, W2 + (size_t)l * FFDIM * DMODEL, DMODEL, w.Wt, FFDIM, FFDIM, DMODEL);
        launch_gemm(stream, FFbf, FFDIM, w.Wt, FFDIM, proj, DMODEL,
                    S_LEN, DMODEL, FFDIM, bf2 + (size_t)l * DMODEL, 0);
        ln_residual<<<S_LEN, 256, 0, stream>>>(proj, t1, g1 + (size_t)l * DMODEL,
                                               b1 + (size_t)l * DMODEL, buf);
    }

    head_softmax<<<1, 1024, 0, stream>>>(buf, Wout, bout, (float*)d_out);
}
